// cross_attn_AWG_3530463117915
// MI455X (gfx1250) — matrix operand order repaired and checked by reading, not compiled, not run on hardware
//
#include <hip/hip_runtime.h>
#include <hip/hip_bf16.h>

// ---------------------------------------------------------------------------
// Cross-attention with AWG top-L gating for MI455X (gfx1250, wave32, WMMA+TDM)
// B=4, N=2048, DMODEL=512, H=8, DK=DV=64, L=32.
// ---------------------------------------------------------------------------

typedef __attribute__((ext_vector_type(16))) _Float16     v16h;
typedef __attribute__((ext_vector_type(8)))  float        v8f;
typedef __attribute__((ext_vector_type(4)))  unsigned int u32x4;
typedef __attribute__((ext_vector_type(8)))  int          i32x8;
typedef __attribute__((ext_vector_type(4)))  int          i32x4;

#define BATCH   4
#define SEQ     2048
#define DMODEL  512
#define NHEAD   8
#define DHEAD   64
#define TOPL    32
#define MAXKEEP 64      // capacity for kept keys per row (ties slack)
#define KCHUNK  64      // keys staged per TDM transfer (64*64*2B = 8 KB)

#if __has_builtin(__builtin_amdgcn_tensor_load_to_lds)
#define HAS_TDM 1
#else
#define HAS_TDM 0
#endif

__device__ __forceinline__ float wred_sum_f(float v) {
#pragma unroll
  for (int m = 16; m > 0; m >>= 1) v += __shfl_xor(v, m, 32);
  return v;
}
__device__ __forceinline__ float wred_max_f(float v) {
#pragma unroll
  for (int m = 16; m > 0; m >>= 1) v = fmaxf(v, __shfl_xor(v, m, 32));
  return v;
}
__device__ __forceinline__ float wred_min_f(float v) {
#pragma unroll
  for (int m = 16; m > 0; m >>= 1) v = fminf(v, __shfl_xor(v, m, 32));
  return v;
}
__device__ __forceinline__ int wred_sum_i(int v) {
#pragma unroll
  for (int m = 16; m > 0; m >>= 1) v += __shfl_xor(v, m, 32);
  return v;
}

// ---------------------------------------------------------------------------
// TDM helper: 1-D async copy of `nelem` f16 elements global -> LDS.
// Descriptor layout per CDNA5 ISA 8.3/8.4 (count=1, type=2, data_size=2B).
// 6-arg builtin form: (g0, g1, g2, g3, g4, cpol).
// ---------------------------------------------------------------------------
__device__ __forceinline__ void tdm_copy_f16(const _Float16* src,
                                             _Float16* dst_lds,
                                             unsigned nelem, int lane) {
#if HAS_TDM
  unsigned long long ga = (unsigned long long)(const void*)src;
  unsigned lds_off = (unsigned)(unsigned long long)(void*)dst_lds;
  u32x4 g0;
  g0.x = 1u;                                            // count=1 (valid)
  g0.y = lds_off;                                       // lds_addr
  g0.z = (unsigned)(ga & 0xFFFFFFFFu);                  // global_addr[31:0]
  g0.w = (unsigned)((ga >> 32) & 0x01FFFFFFu) | (2u << 30);  // addr hi | type=2
  i32x8 g1 = {0, 0, 0, 0, 0, 0, 0, 0};
  g1[0] = (1 << 16);                                    // data_size = 2 bytes
  g1[1] = (int)((nelem & 0xFFFFu) << 16);               // tensor_dim0[15:0]
  g1[2] = (int)((nelem >> 16) & 0xFFFFu) | (1 << 16);   // td0 hi | tensor_dim1=1
  g1[3] = (int)((nelem & 0xFFFFu) << 16);               // tile_dim0 = nelem
  g1[4] = 1;                                            // tile_dim1 = 1
  g1[5] = (int)nelem;                                   // tensor_dim0_stride lo
  i32x4 z4 = {0, 0, 0, 0};
  i32x8 z8 = {0, 0, 0, 0, 0, 0, 0, 0};
  __builtin_amdgcn_tensor_load_to_lds(g0, g1, z4, z4, z8, 0);
#else
  const u32x4* s = (const u32x4*)src;
  u32x4* d = (u32x4*)dst_lds;
  const int n16 = (int)nelem / 8;                       // 8 halfs per 16B op
  for (int i = lane; i < n16; i += 32) d[i] = s[i];
#endif
}
template <int N>
__device__ __forceinline__ void tdm_wait() {
#if HAS_TDM
  __builtin_amdgcn_s_wait_tensorcnt(N);                 // N is an ICE
#endif
}
__device__ __forceinline__ void lds_read_fence() {
#if HAS_TDM
  asm volatile("s_wait_dscnt 0x0" ::: "memory");        // buffer-reuse fence
#endif
}

// ---------------------------------------------------------------------------
// Kernel 0: weight convert+transpose: W[512x512] f32 row-major ->
// WT[512x512] f16, WT[c][k] = W[k][c]  (makes WMMA B-fragments contiguous).
// ---------------------------------------------------------------------------
__global__ void convert_wT_kernel(const float* __restrict__ W,
                                  _Float16* __restrict__ WT) {
  const int idx = blockIdx.x * 256 + threadIdx.x;       // over 512*512
  const int k = idx >> 9;
  const int c = idx & (DMODEL - 1);
  WT[(size_t)c * DMODEL + k] = (_Float16)W[idx];
}

// ---------------------------------------------------------------------------
// Kernel 1: projection GEMM  Y = X[8192x512] @ W, f16 out, head-major
// [B][H][N][DHEAD].  One wave per 16(M)x64(N) tile: 4 accumulators share A.
// grid = (DMODEL/64, B*SEQ/16), block = 32.
// ---------------------------------------------------------------------------
__global__ void proj_wmma_kernel(const float* __restrict__ X,
                                 const _Float16* __restrict__ WT,
                                 _Float16* __restrict__ Y) {
  const int lane = threadIdx.x & 31;
  const int n0 = blockIdx.x * 64;
  const int m0 = blockIdx.y * 16;
  const int arow = m0 + (lane & 15);

  v8f acc[4];
#pragma unroll
  for (int t = 0; t < 4; ++t)
#pragma unroll
    for (int i = 0; i < 8; ++i) acc[t][i] = 0.0f;

  for (int kk = 0; kk < DMODEL; kk += 32) {
    v16h a;
    const int kaBase = kk + ((lane < 16) ? 0 : 16);
#pragma unroll
    for (int j = 0; j < 8; ++j) {
      a[j]     = (_Float16)X[(size_t)arow * DMODEL + kaBase + j];
      a[j + 8] = (_Float16)X[(size_t)arow * DMODEL + kaBase + 8 + j];
    }
    const int kbBase = kk + ((lane < 16) ? 0 : 16);
#pragma unroll
    for (int nt = 0; nt < 4; ++nt) {
      const int bcol = n0 + nt * 16 + (lane & 15);
      v16h bf;
#pragma unroll
      for (int j = 0; j < 16; ++j)
        bf[j] = WT[(size_t)bcol * DMODEL + kbBase + j];  // contiguous 32B
      acc[nt] = __builtin_amdgcn_wmma_f32_16x16x32_f16(false, a, false, bf,
                                                       (short)0, acc[nt],
                                                       false, false);
    }
  }

  const int mbase = m0 + ((lane < 16) ? 0 : 8);
#pragma unroll
  for (int nt = 0; nt < 4; ++nt) {
    const int col = n0 + nt * 16 + (lane & 15);
    const int h = col >> 6;
    const int d = col & 63;
#pragma unroll
    for (int v = 0; v < 8; ++v) {
      const int m = mbase + v;
      const int b = m >> 11;
      const int n = m & (SEQ - 1);
      Y[(((size_t)b * NHEAD + h) * SEQ + n) * DHEAD + d] = (_Float16)acc[nt][v];
    }
  }
}

// ---------------------------------------------------------------------------
// Kernel 2: attention. One wave per (b, h, 16-query tile).
// K panel streamed into LDS in 8KB chunks by the Tensor Data Mover (double
// buffered, s_wait_tensorcnt). Scores 16x2048 f32 in LDS. Top-32 threshold
// via binary search, ballot compaction, sparse w*V gather, renormalize.
// grid = (SEQ/16, NHEAD, BATCH), block = 32,
// dynLDS = 16*SEQ*4 + 2*KCHUNK*DHEAD*2 = 147456 B.
// ---------------------------------------------------------------------------
__global__ void attn_awg_kernel(const _Float16* __restrict__ Qh,
                                const _Float16* __restrict__ Kh,
                                const _Float16* __restrict__ Vh,
                                _Float16* __restrict__ AO) {
  extern __shared__ char smem[];
  float* sc = (float*)smem;                               // [16][SEQ]
  _Float16* kbuf = (_Float16*)(smem + 16 * SEQ * sizeof(float)); // [2][KCHUNK*64]
  __shared__ int   kidx[MAXKEEP];
  __shared__ float kw[MAXKEEP];

  const int lane = threadIdx.x & 31;
  const int q0 = blockIdx.x * 16;
  const int h  = blockIdx.y;
  const int b  = blockIdx.z;
  const size_t base = ((size_t)(b * NHEAD + h)) * SEQ * DHEAD;
  const _Float16* Kb = Kh + base;

  // ---- Q A-fragments for the two K-steps (DK = 64 = 2 x 32) ----
  v16h aq[2];
  const int qrow = q0 + (lane & 15);
#pragma unroll
  for (int t = 0; t < 2; ++t) {
    const int kaBase = t * 32 + ((lane < 16) ? 0 : 16);
#pragma unroll
    for (int j = 0; j < 8; ++j) {
      aq[t][j]     = Qh[base + (size_t)qrow * DHEAD + kaBase + j];
      aq[t][j + 8] = Qh[base + (size_t)qrow * DHEAD + kaBase + 8 + j];
    }
  }

  // ---- scores: S = (Q/sqrt(64)) K^T; TDM-staged K chunks ----
  const float scale = 0.125f;
  const int NCH = SEQ / KCHUNK;
  tdm_copy_f16(Kb, kbuf, KCHUNK * DHEAD, lane);
  for (int c = 0; c < NCH; ++c) {
    if (c + 1 < NCH) {
      lds_read_fence();   // chunk c-1 reads retired before overwriting its buf
      tdm_copy_f16(Kb + (size_t)(c + 1) * KCHUNK * DHEAD,
                   kbuf + ((c + 1) & 1) * (KCHUNK * DHEAD),
                   KCHUNK * DHEAD, lane);
      tdm_wait<1>();      // chunk c resident
    } else {
      tdm_wait<0>();
    }
    const _Float16* kb = kbuf + (c & 1) * (KCHUNK * DHEAD);
#pragma unroll
    for (int t16 = 0; t16 < KCHUNK / 16; ++t16) {
      v8f cacc;
#pragma unroll
      for (int i = 0; i < 8; ++i) cacc[i] = 0.0f;
      const int lkey = t16 * 16 + (lane & 15);
#pragma unroll
      for (int t = 0; t < 2; ++t) {
        const int kbBase = t * 32 + ((lane < 16) ? 0 : 16);
        v16h bf;
#pragma unroll
        for (int j = 0; j < 16; ++j) bf[j] = kb[lkey * DHEAD + kbBase + j];
        cacc = __builtin_amdgcn_wmma_f32_16x16x32_f16(false, aq[t], false, bf,
                                                      (short)0, cacc,
                                                      false, false);
      }
      const int mbase = (lane < 16) ? 0 : 8;
      const int ncol  = c * KCHUNK + t16 * 16 + (lane & 15);
#pragma unroll
      for (int v = 0; v < 8; ++v) sc[(mbase + v) * SEQ + ncol] = cacc[v] * scale;
    }
  }
  __builtin_amdgcn_wave_barrier();   // order LDS score writes vs reads below

  // ---- per-row: top-32 threshold, gate, renormalize, sparse PV ----
  const _Float16* Vb = Vh + base;
  const int d0 = lane * 2;

  for (int r = 0; r < 16; ++r) {
    float* row = sc + r * SEQ;
    float vloc[SEQ / 32];
#pragma unroll
    for (int i = 0; i < SEQ / 32; ++i) vloc[i] = row[i * 32 + lane];

    float mx = -3.4e38f, mn = 3.4e38f;
#pragma unroll
    for (int i = 0; i < SEQ / 32; ++i) {
      mx = fmaxf(mx, vloc[i]);
      mn = fminf(mn, vloc[i]);
    }
    mx = wred_max_f(mx);
    mn = wred_min_f(mn);

    // binary search for 32nd-largest logit: invariant count(lo) >= TOPL
    float lo = mn, hi = mx;
    for (int it = 0; it < 32; ++it) {
      const float mid = 0.5f * (lo + hi);
      int cnt = 0;
#pragma unroll
      for (int i = 0; i < SEQ / 32; ++i) cnt += (vloc[i] >= mid) ? 1 : 0;
      cnt = wred_sum_i(cnt);
      if (cnt >= TOPL) lo = mid; else hi = mid;
    }
    const float thr = lo;

    // compaction of surviving keys + softmax weights
    int cnt = 0;
    float wsumLocal = 0.0f;
#pragma unroll
    for (int ch = 0; ch < SEQ / 32; ++ch) {
      const float s = vloc[ch];
      const bool keep = (s >= thr);
      const unsigned mask = (unsigned)__ballot(keep);
      const int pos = cnt + __popc(mask & ((1u << lane) - 1u));
      if (keep && pos < MAXKEEP) {
        const float w = __expf(s - mx);
        kidx[pos] = ch * 32 + lane;
        kw[pos]   = w;
        wsumLocal += w;
      }
      cnt += __popc(mask);
    }
    const int ncap = (cnt < MAXKEEP) ? cnt : MAXKEEP;
    const float denom = wred_sum_f(wsumLocal);
    __builtin_amdgcn_wave_barrier();

    // sparse P*V: each lane owns 2 output dims
    float acc0 = 0.0f, acc1 = 0.0f;
    for (int i = 0; i < ncap; ++i) {
      const float w = kw[i];
      const int key = kidx[i];
      acc0 += w * (float)Vb[(size_t)key * DHEAD + d0];
      acc1 += w * (float)Vb[(size_t)key * DHEAD + d0 + 1];
    }
    const float inv = 1.0f / denom;
    const size_t orow = ((size_t)b * SEQ + (q0 + r)) * (NHEAD * DHEAD) + h * DHEAD;
    AO[orow + d0]     = (_Float16)(acc0 * inv);
    AO[orow + d0 + 1] = (_Float16)(acc1 * inv);
    __builtin_amdgcn_wave_barrier();  // kidx/kw reused next row
  }
}

// ---------------------------------------------------------------------------
// Kernel 3: out = LN(AO @ Wfc + residual).  4 waves per 16-row block; each
// wave covers 8 of the 32 column tiles via WMMA, then wave-cooperative LN.
// grid = (B*SEQ/16), block = 128.
// ---------------------------------------------------------------------------
__global__ void fc_res_ln_kernel(const _Float16* __restrict__ AO,
                                 const _Float16* __restrict__ WfcT,
                                 const float* __restrict__ Qres,
                                 const float* __restrict__ gamma,
                                 const float* __restrict__ beta,
                                 float* __restrict__ out) {
  __shared__ float buf[16 * DMODEL];
  const int lane = threadIdx.x & 31;
  const int wave = threadIdx.x >> 5;
  const int m0 = blockIdx.x * 16;
  const int arow = m0 + (lane & 15);

  for (int ct = wave; ct < DMODEL / 16; ct += 4) {
    const int n0 = ct * 16;
    v8f c;
#pragma unroll
    for (int i = 0; i < 8; ++i) c[i] = 0.0f;
    for (int kk = 0; kk < DMODEL; kk += 32) {
      v16h a, bf;
      const int kaBase = kk + ((lane < 16) ? 0 : 16);
#pragma unroll
      for (int j = 0; j < 8; ++j) {
        a[j]     = AO[(size_t)arow * DMODEL + kaBase + j];
        a[j + 8] = AO[(size_t)arow * DMODEL + kaBase + 8 + j];
      }
      const int kbBase = kk + ((lane < 16) ? 0 : 16);
      const int bcol = n0 + (lane & 15);
#pragma unroll
      for (int j = 0; j < 16; ++j)
        bf[j] = WfcT[(size_t)bcol * DMODEL + kbBase + j];  // contiguous 32B
      c = __builtin_amdgcn_wmma_f32_16x16x32_f16(false, a, false, bf,
                                                 (short)0, c, false, false);
    }
    const int mbase = (lane < 16) ? 0 : 8;
    const int col = n0 + (lane & 15);
#pragma unroll
    for (int v = 0; v < 8; ++v) buf[(mbase + v) * DMODEL + col] = c[v];
  }
  __syncthreads();

  for (int r = wave; r < 16; r += 4) {
    const int gm = m0 + r;
    float vals[DMODEL / 32];
    float s = 0.0f, s2 = 0.0f;
#pragma unroll
    for (int i = 0; i < DMODEL / 32; ++i) {
      const int cdx = lane + i * 32;
      const float v = buf[r * DMODEL + cdx] + Qres[(size_t)gm * DMODEL + cdx];
      vals[i] = v;
      s += v;
      s2 += v * v;
    }
    s = wred_sum_f(s);
    s2 = wred_sum_f(s2);
    const float mean = s * (1.0f / DMODEL);
    const float var = s2 * (1.0f / DMODEL) - mean * mean;
    const float rstd = rsqrtf(var + 1e-6f);
#pragma unroll
    for (int i = 0; i < DMODEL / 32; ++i) {
      const int cdx = lane + i * 32;
      out[(size_t)gm * DMODEL + cdx] =
          (vals[i] - mean) * rstd * gamma[cdx] + beta[cdx];
    }
  }
}

// ---------------------------------------------------------------------------
extern "C" void kernel_launch(void* const* d_in, const int* in_sizes, int n_in,
                              void* d_out, int out_size, void* d_ws, size_t ws_size,
                              hipStream_t stream) {
  const float* q    = (const float*)d_in[0];
  const float* k    = (const float*)d_in[1];
  const float* v    = (const float*)d_in[2];
  const float* Wq   = (const float*)d_in[3];
  const float* Wk   = (const float*)d_in[4];
  const float* Wv   = (const float*)d_in[5];
  const float* Wfc  = (const float*)d_in[6];
  const float* ln_g = (const float*)d_in[7];
  const float* ln_b = (const float*)d_in[8];

  const size_t elems = (size_t)BATCH * SEQ * DMODEL;   // 4,194,304
  const size_t welems = (size_t)DMODEL * DMODEL;       //   262,144
  _Float16* Qh   = (_Float16*)d_ws;
  _Float16* Kh   = Qh + elems;
  _Float16* Vh   = Kh + elems;
  _Float16* AO   = Vh + elems;
  _Float16* WqT  = AO + elems;
  _Float16* WkT  = WqT + welems;
  _Float16* WvT  = WkT + welems;
  _Float16* WfcT = WvT + welems;

  // 0) weight convert + transpose (f32 row-major -> f16 col-major)
  {
    dim3 grid(welems / 256);
    convert_wT_kernel<<<grid, 256, 0, stream>>>(Wq,  WqT);
    convert_wT_kernel<<<grid, 256, 0, stream>>>(Wk,  WkT);
    convert_wT_kernel<<<grid, 256, 0, stream>>>(Wv,  WvT);
    convert_wT_kernel<<<grid, 256, 0, stream>>>(Wfc, WfcT);
  }
  // 1) projections
  {
    dim3 grid(DMODEL / 64, (BATCH * SEQ) / 16);
    proj_wmma_kernel<<<grid, 32, 0, stream>>>(q, WqT, Qh);
    proj_wmma_kernel<<<grid, 32, 0, stream>>>(k, WkT, Kh);
    proj_wmma_kernel<<<grid, 32, 0, stream>>>(v, WvT, Vh);
  }
  // 2) attention with top-L gating (TDM-staged K panel)
  {
    dim3 grid(SEQ / 16, NHEAD, BATCH);
    const size_t lds = (size_t)16 * SEQ * sizeof(float)
                     + (size_t)2 * KCHUNK * DHEAD * sizeof(_Float16);
    attn_awg_kernel<<<grid, 32, lds, stream>>>(Qh, Kh, Vh, AO);
  }
  // 3) FC + residual + LayerNorm
  {
    dim3 grid((BATCH * SEQ) / 16);
    fc_res_ln_kernel<<<grid, 128, 0, stream>>>(AO, WfcT, q, ln_g, ln_b,
                                               (float*)d_out);
  }
}
